// RNNCute_35648228557372
// MI455X (gfx1250) — matrix operand order repaired and checked by reading, not compiled, not run on hardware
//
#include <hip/hip_runtime.h>
#include <hip/hip_bf16.h>

// ---------------------------------------------------------------------------
// MI455X / gfx1250:
//   u = x @ w_in + b_in        WMMA f16 GEMM (f32 accum), async Global->LDS
//   h_s = tanh(h_{s-1}@W_n+u)  W_n resident in VGPRs as WMMA B-fragments
//   out = y @ w_out            WMMA f16 GEMM
// Weights are converted f32->f16 AND transposed once, so GEMM B-tiles are
// contiguous-K row copies (b128 / async-to-LDS friendly, no LDS scatter).
// ---------------------------------------------------------------------------

typedef _Float16 hvec4  __attribute__((ext_vector_type(4)));
typedef _Float16 hvec8  __attribute__((ext_vector_type(8)));
typedef _Float16 hvec16 __attribute__((ext_vector_type(16)));
typedef float    fvec8  __attribute__((ext_vector_type(8)));
typedef int      ivec4  __attribute__((ext_vector_type(4)));

__device__ __forceinline__ hvec16 cat8(hvec8 lo, hvec8 hi) {
  hvec16 r;
#pragma unroll
  for (int e = 0; e < 8; ++e) { r[e] = lo[e]; r[8 + e] = hi[e]; }
  return r;
}

// -------- CDNA5 async Global->LDS copy (guarded; falls back to manual) -----
#if __has_builtin(__builtin_amdgcn_global_load_async_to_lds_b128)
#define HAS_ASYNC_LDS 1
__device__ __forceinline__ void async_b128(const _Float16* g, _Float16* l) {
  __builtin_amdgcn_global_load_async_to_lds_b128(
      (__attribute__((address_space(1))) ivec4*)(g),
      (__attribute__((address_space(3))) ivec4*)(l), 0, 0);
}
#if __has_builtin(__builtin_amdgcn_s_wait_asynccnt)
#define WAIT_ASYNC(n) __builtin_amdgcn_s_wait_asynccnt(n)
#else
#define WAIT_ASYNC(n) asm volatile("s_wait_asynccnt %0" ::"n"(n) : "memory")
#endif
#else
#define HAS_ASYNC_LDS 0
#endif

// ---------------- f32 -> f16 bulk convert (n multiple of 8) ----------------
__global__ __launch_bounds__(256) void cvt_f32_to_f16(const float* __restrict__ src,
                                                      _Float16* __restrict__ dst,
                                                      int n) {
  int i = (blockIdx.x * 256 + threadIdx.x) * 8;
  if (i < n) {
    float4 a = *(const float4*)(src + i);
    float4 b = *(const float4*)(src + i + 4);
    hvec8 o;
    o[0] = (_Float16)a.x; o[1] = (_Float16)a.y; o[2] = (_Float16)a.z; o[3] = (_Float16)a.w;
    o[4] = (_Float16)b.x; o[5] = (_Float16)b.y; o[6] = (_Float16)b.z; o[7] = (_Float16)b.w;
    *(hvec8*)(dst + i) = o;
  }
}

// -------- f32 -> f16 convert + transpose: dst[C][R] = (f16) src[R][C] ------
// R, C multiples of 64. 64x64 tiles via LDS.
__global__ __launch_bounds__(256) void cvt_t_f32_to_f16(const float* __restrict__ src,
                                                        _Float16* __restrict__ dst,
                                                        int R, int Cc) {
  __shared__ __align__(16) float tile[64][68];
  const int t = threadIdx.x;
  const int r0 = blockIdx.y * 64, c0 = blockIdx.x * 64;
#pragma unroll
  for (int i = 0; i < 4; ++i) {
    int row = t >> 2;
    int seg = (t & 3) * 4 + i;
    float4 v = *(const float4*)(src + (size_t)(r0 + row) * Cc + c0 + seg * 4);
    *(float4*)&tile[row][seg * 4] = v;
  }
  __syncthreads();
#pragma unroll
  for (int i = 0; i < 4; ++i) {
    int col = t >> 2;
    int seg = (t & 3) * 4 + i;
    hvec4 o;
#pragma unroll
    for (int e = 0; e < 4; ++e) o[e] = (_Float16)tile[seg * 4 + e][col];
    *(hvec4*)(dst + (size_t)(c0 + col) * R + r0 + seg * 4) = o;
  }
}

// ---------------- tiled WMMA GEMM: C[M,N] = A[M,K] * BT[N,K]^T (+bias) -----
// A: f16 [M][K]; BT: f16 [N][K] (pre-transposed); C: f32 [M][N].
#define BM 128
#define BN 128
#define BK 32
#define LDA 56   // padded LDS row stride (halves): 112B, 16B aligned, conflict-free

__global__ __launch_bounds__(256) void gemm_f16_wmma(const _Float16* __restrict__ A,
                                                     const _Float16* __restrict__ BT,
                                                     const float* __restrict__ bias,
                                                     float* __restrict__ C,
                                                     int M, int N, int K) {
  __shared__ __align__(16) _Float16 As[2][BM * LDA];  // double buffered, 14 KB each
  __shared__ __align__(16) _Float16 Bs[2][BN * LDA];

  const int tid  = threadIdx.x;
  const int lane = tid & 31;
  const int wid  = tid >> 5;
  const int wm   = (wid & 3) * 32;   // 4 waves tile M
  const int wn   = (wid >> 2) * 64;  // 2 waves tile N
  const int g    = lane >> 4;
  const int lc   = lane & 15;

  const int mBase = blockIdx.y * BM;
  const int nBase = blockIdx.x * BN;

  // per-thread staging coordinates: 512 16B segments per tile, 2 per thread
  int sr[2], sc[2];
#pragma unroll
  for (int i = 0; i < 2; ++i) {
    int sid = tid * 2 + i;
    sr[i] = sid >> 2;          // row 0..127
    sc[i] = (sid & 3) * 8;     // k offset 0/8/16/24
  }

  fvec8 acc[2][4];
#pragma unroll
  for (int i = 0; i < 2; ++i)
#pragma unroll
    for (int j = 0; j < 4; ++j)
#pragma unroll
      for (int r = 0; r < 8; ++r) acc[i][j][r] = 0.0f;

  auto compute = [&](int b) {
    hvec16 af[2];
#pragma unroll
    for (int mf = 0; mf < 2; ++mf) {
      const _Float16* p = &As[b][(wm + mf * 16 + lc) * LDA + g * 8];
      af[mf] = cat8(*(const hvec8*)p, *(const hvec8*)(p + 16));
    }
    hvec16 bfv[4];
#pragma unroll
    for (int nf = 0; nf < 4; ++nf) {
      const _Float16* p = &Bs[b][(wn + nf * 16 + lc) * LDA + g * 8];
      bfv[nf] = cat8(*(const hvec8*)p, *(const hvec8*)(p + 16));
    }
#pragma unroll
    for (int mf = 0; mf < 2; ++mf)
#pragma unroll
      for (int nf = 0; nf < 4; ++nf)
        acc[mf][nf] = __builtin_amdgcn_wmma_f32_16x16x32_f16(
            false, af[mf], false, bfv[nf], (short)0, acc[mf][nf], false, false);
  };

  const int nIter = K / BK;

#if HAS_ASYNC_LDS
  auto stage = [&](int kt, int b) {
    int k0 = kt * BK;
#pragma unroll
    for (int i = 0; i < 2; ++i) {
      async_b128(A  + (size_t)(mBase + sr[i]) * K + k0 + sc[i], &As[b][sr[i] * LDA + sc[i]]);
      async_b128(BT + (size_t)(nBase + sr[i]) * K + k0 + sc[i], &Bs[b][sr[i] * LDA + sc[i]]);
    }
  };
  stage(0, 0);
  for (int it = 0; it < nIter; ++it) {
    int cur = it & 1;
    if (it + 1 < nIter) {           // DMA next tile while this one computes
      stage(it + 1, cur ^ 1);
      WAIT_ASYNC(4);                // our 4 tile-`it` transfers are done
    } else {
      WAIT_ASYNC(0);
    }
    __syncthreads();                // all waves' tile-`it` data in LDS
    compute(cur);
    __syncthreads();                // all reads done before buffer reuse
  }
#else
  // register-pipelined fallback: issue next tile's loads before computing
  hvec8 ra[2], rb[2];
#pragma unroll
  for (int i = 0; i < 2; ++i) {
    ra[i] = *(const hvec8*)(A  + (size_t)(mBase + sr[i]) * K + sc[i]);
    rb[i] = *(const hvec8*)(BT + (size_t)(nBase + sr[i]) * K + sc[i]);
  }
  for (int it = 0; it < nIter; ++it) {
    int cur = it & 1;
#pragma unroll
    for (int i = 0; i < 2; ++i) {
      *(hvec8*)(&As[cur][sr[i] * LDA + sc[i]]) = ra[i];
      *(hvec8*)(&Bs[cur][sr[i] * LDA + sc[i]]) = rb[i];
    }
    if (it + 1 < nIter) {
      int k0 = (it + 1) * BK;
#pragma unroll
      for (int i = 0; i < 2; ++i) {
        ra[i] = *(const hvec8*)(A  + (size_t)(mBase + sr[i]) * K + k0 + sc[i]);
        rb[i] = *(const hvec8*)(BT + (size_t)(nBase + sr[i]) * K + k0 + sc[i]);
      }
    }
    __syncthreads();
    compute(cur);
    __syncthreads();
  }
#endif

  // epilogue: C fragment -> row = g*8 + r, col = lane%16
#pragma unroll
  for (int nf = 0; nf < 4; ++nf) {
    int col = nBase + wn + nf * 16 + lc;
    float bv = (bias != nullptr) ? bias[col] : 0.0f;
#pragma unroll
    for (int mf = 0; mf < 2; ++mf)
#pragma unroll
      for (int r = 0; r < 8; ++r) {
        int row = mBase + wm + mf * 16 + g * 8 + r;
        C[(size_t)row * N + col] = acc[mf][nf][r] + bv;
      }
  }
}

// ---------------- recurrence: one wave per head; W_n resident as B-frags ---
__global__ __launch_bounds__(32) void rnn_scan_wmma(const float* __restrict__ u,
                                                    const float* __restrict__ Wst,
                                                    _Float16* __restrict__ y,
                                                    int S) {
  __shared__ __align__(16) float    Wl[64 * 64];
  __shared__ __align__(16) _Float16 st[16 * 72];  // padded stride 72 (144B)

  const int lane = threadIdx.x;
  const int n  = blockIdx.x;
  const int g  = lane >> 4;
  const int lc = lane & 15;

  const float* Wn = Wst + (size_t)n * 4096;
  for (int i = lane; i < 1024; i += 32)
    ((float4*)Wl)[i] = ((const float4*)Wn)[i];
  for (int i = lane; i < 16 * 72; i += 32) st[i] = (_Float16)0.0f;  // rows 4..15 stay 0
  __syncthreads();

  hvec16 bf[4][2];  // resident W_n fragments: cols t*16..+15, K c*32..+31
#pragma unroll
  for (int t = 0; t < 4; ++t)
#pragma unroll
    for (int c = 0; c < 2; ++c) {
      hvec16 b;
#pragma unroll
      for (int j = 0; j < 16; ++j) {
        int kk = c * 32 + (j < 8 ? g * 8 + j : 16 + g * 8 + (j - 8));
        b[j] = (_Float16)Wl[kk * 64 + t * 16 + lc];
      }
      bf[t][c] = b;
    }
  __syncthreads();

  hvec16 a0, a1;
#pragma unroll
  for (int j = 0; j < 16; ++j) { a0[j] = (_Float16)0.0f; a1[j] = (_Float16)0.0f; }

  for (int s = 0; s < S; ++s) {
    if (s + 8 < S) {  // gfx1250 prefetch of upcoming u rows
#pragma unroll
      for (int r = 0; r < 4; ++r)
        __builtin_prefetch(u + (size_t)(r * S + s + 8) * 2048 + n * 64 + lane * 2, 0, 1);
    }
    _Float16 ov[4][4];
#pragma unroll
    for (int t = 0; t < 4; ++t) {
      fvec8 c;
#pragma unroll
      for (int r = 0; r < 8; ++r) c[r] = 0.0f;
      if (g == 0) {
#pragma unroll
        for (int r = 0; r < 4; ++r)
          c[r] = u[(size_t)(r * S + s) * 2048 + n * 64 + t * 16 + lc];
      }
      c = __builtin_amdgcn_wmma_f32_16x16x32_f16(false, a0, false, bf[t][0], (short)0, c, false, false);
      c = __builtin_amdgcn_wmma_f32_16x16x32_f16(false, a1, false, bf[t][1], (short)0, c, false, false);
#pragma unroll
      for (int r = 0; r < 4; ++r) ov[t][r] = (_Float16)tanhf(c[r]);
    }
    if (g == 0) {
#pragma unroll
      for (int t = 0; t < 4; ++t)
#pragma unroll
        for (int r = 0; r < 4; ++r) {
          st[r * 72 + t * 16 + lc] = ov[t][r];
          y[(size_t)(r * S + s) * 2048 + n * 64 + t * 16 + lc] = ov[t][r];
        }
    }
    __syncthreads();  // single-wave WG: NOP barrier; LDS in-order per wave
    {
      const _Float16* p = &st[lc * 72 + g * 8];
      a0 = cat8(*(const hvec8*)p, *(const hvec8*)(p + 16));
      const _Float16* q = &st[lc * 72 + 32 + g * 8];
      a1 = cat8(*(const hvec8*)q, *(const hvec8*)(q + 16));
    }
  }
}

// ---------------------------------------------------------------------------
extern "C" void kernel_launch(void* const* d_in, const int* in_sizes, int n_in,
                              void* d_out, int out_size, void* d_ws, size_t ws_size,
                              hipStream_t stream) {
  (void)in_sizes; (void)n_in; (void)out_size; (void)ws_size;
  const float* x     = (const float*)d_in[0];   // [4,2048,2048]
  const float* w_in  = (const float*)d_in[1];   // [2048,2048]
  const float* b_in  = (const float*)d_in[2];   // [2048]
  const float* w_st  = (const float*)d_in[3];   // [32,64,64]
  const float* w_out = (const float*)d_in[4];   // [2048,2048]
  float* out = (float*)d_out;                   // [8192,2048]

  const int S = 2048, D = 2048, Nh = 32, O = 2048;
  const int M = 4 * S;   // 8192
  const int NH = 2048;
  const size_t xe = (size_t)M * D;
  const size_t we = (size_t)D * NH;
  const size_t oe = (size_t)NH * O;

  char* ws = (char*)d_ws;
  _Float16* xh     = (_Float16*)ws;  ws += xe * 2;             // 32 MB
  _Float16* winhT  = (_Float16*)ws;  ws += we * 2;             //  8 MB  [NH][D]
  _Float16* wouthT = (_Float16*)ws;  ws += oe * 2;             //  8 MB  [O][NH]
  float*    u      = (float*)ws;     ws += (size_t)M * NH * 4; // 64 MB
  _Float16* yh     = (_Float16*)ws;                            // 32 MB

  cvt_f32_to_f16<<<(int)(xe / 8 / 256), 256, 0, stream>>>(x, xh, (int)xe);
  cvt_t_f32_to_f16<<<dim3(NH / 64, D / 64), 256, 0, stream>>>(w_in, winhT, D, NH);
  cvt_t_f32_to_f16<<<dim3(O / 64, NH / 64), 256, 0, stream>>>(w_out, wouthT, NH, O);

  gemm_f16_wmma<<<dim3(NH / BN, M / BM), 256, 0, stream>>>(xh, winhT, b_in, u, M, NH, D);
  rnn_scan_wmma<<<Nh, 32, 0, stream>>>(u, w_st, yh, S);
  gemm_f16_wmma<<<dim3(O / BN, M / BM), 256, 0, stream>>>(yh, wouthT, nullptr, out, M, O, NH);
}
